// ClusteringLayer_71279277244876
// MI455X (gfx1250) — compile-verified, matrix operand order checked by reading, not run on hardware
//
#include <hip/hip_runtime.h>

typedef __attribute__((ext_vector_type(16))) __bf16 v16bf;
typedef __attribute__((ext_vector_type(8)))  float  v8f;
typedef __attribute__((ext_vector_type(4)))  int    v4i;

#define DIM   512      // feature dim
#define KC    512      // number of centroids
#define NROWS 65536
#define BLK_M 128      // rows per workgroup
#define NWAVE 16       // 512 threads

#if __has_builtin(__builtin_amdgcn_global_load_async_to_lds_b128) && \
    __has_builtin(__builtin_amdgcn_s_wait_asynccnt)
#define HAS_ASYNC 1
#else
#define HAS_ASYNC 0
#endif

#if __has_builtin(__builtin_amdgcn_sched_barrier)
#define SCHED_FENCE() __builtin_amdgcn_sched_barrier(0)
#else
#define SCHED_FENCE()
#endif

#if HAS_ASYNC
typedef __attribute__((address_space(1))) v4i gv4i;   // global int4
typedef __attribute__((address_space(3))) v4i lv4i;   // LDS int4
__device__ __forceinline__ void xasync16(const float* g, void* l) {
    __builtin_amdgcn_global_load_async_to_lds_b128(
        (gv4i*)const_cast<float*>(g), (lv4i*)l, 0, 0);
}
#endif

// d_ws layout:
//   [0, 1MB)        : centroid B-fragments, split bf16
//                     byte offset(ctg, dc, h, lane) = ctg*32768 + dc*2048 + h*1024 + lane*32
//   [1MB, 1MB+2KB)  : csq[512] = ||c_k||^2 (f32)

// ---------------------------------------------------------------------------
// Prep 1: centroids fp32 -> split-bf16 WMMA B fragments (pre-swizzled).
//   lane n (0..15): col N=n, K = kchunk*32 + {0..7, 16..23}
//   lane n+16     : col N=n, K = kchunk*32 + {8..15, 24..31}
// ---------------------------------------------------------------------------
__global__ void prep_b(const float* __restrict__ c, char* __restrict__ wsB) {
    int ctg  = blockIdx.x >> 4;          // 0..31
    int dc   = blockIdx.x & 15;          // 0..15
    int lane = threadIdx.x;              // 0..31
    int n     = ctg * 16 + (lane & 15);
    int kbase = dc * 32 + ((lane & 16) ? 8 : 0);
    const float* cp = c + (size_t)n * DIM + kbase;

    float f[16];
#pragma unroll
    for (int i = 0; i < 8; ++i) f[i]     = cp[i];
#pragma unroll
    for (int i = 0; i < 8; ++i) f[8 + i] = cp[16 + i];

    v16bf hi, lo;
#pragma unroll
    for (int i = 0; i < 16; ++i) {
        __bf16 h = (__bf16)f[i];
        hi[i] = h;
        lo[i] = (__bf16)(f[i] - (float)h);
    }
    size_t base = (size_t)ctg * 32768 + (size_t)dc * 2048 + (size_t)lane * 32;
    *(v16bf*)(wsB + base)        = hi;
    *(v16bf*)(wsB + base + 1024) = lo;
}

// ---------------------------------------------------------------------------
// Prep 2: csq[k] = sum_d c[k][d]^2   (one wave per centroid row)
// ---------------------------------------------------------------------------
__global__ void prep_csq(const float* __restrict__ c, float* __restrict__ csq) {
    int wave = threadIdx.x >> 5;
    int lane = threadIdx.x & 31;
    int row  = blockIdx.x * 8 + wave;
    const float* cp = c + (size_t)row * DIM;
    float s = 0.f;
#pragma unroll
    for (int k = lane * 4; k < DIM; k += 128) {
        float4 v = *(const float4*)(cp + k);
        s += v.x * v.x + v.y * v.y + v.z * v.z + v.w * v.w;
    }
#pragma unroll
    for (int m = 16; m >= 1; m >>= 1) s += __shfl_xor(s, m);
    if (lane == 0) csq[row] = s;
}

// ---------------------------------------------------------------------------
// Main kernel: 512 threads = 16 waves; block = 128 rows x 512 centroids.
// Wave w: row-group rg = w>>1 (16 rows), col-group cg = w&1 (256 cols, 16 tiles).
// x stream: ASYNCcnt-decoupled global->LDS double buffer.
// B stream: L2-resident pre-swizzled fragments; explicit 2-stage rotated
//           pipeline over tile pairs with sched_barrier fences so at most
//           2 pairs (64 VGPRs) of fragments are in flight -> no spills.
// ---------------------------------------------------------------------------
__global__ void __launch_bounds__(512)
qkernel(const float* __restrict__ x, const char* __restrict__ wsB,
        const float* __restrict__ csq, float* __restrict__ out) {
    __shared__ float rowsum[BLK_M];
#if HAS_ASYNC
    __shared__ __align__(16) char xstage[NWAVE * 2 * 2560];  // 80 KB, per-wave dbl buf
#endif

    const int tid    = threadIdx.x;
    const int w      = tid >> 5;          // wave 0..15
    const int rg     = w >> 1;            // row-group 0..7
    const int cg     = w & 1;             // col-group 0..1
    const int lane   = tid & 31;
    const int laneHi = (lane >> 4) & 1;
    const int ln     = lane & 15;
    const int blockRow = blockIdx.x * BLK_M;

    if (tid < BLK_M) rowsum[tid] = 0.f;
    __syncthreads();

    v8f acc[16];
#pragma unroll
    for (int ct = 0; ct < 16; ++ct)
#pragma unroll
        for (int j = 0; j < 8; ++j) acc[ct][j] = 0.f;

    float xsq = 0.f;

    const float* xrow = x + (size_t)(blockRow + rg * 16 + ln) * DIM + laneHi * 8;
    const char*  bp   = wsB + (size_t)cg * 524288 + (size_t)lane * 32;

#if HAS_ASYNC
    char* mybuf = xstage + (size_t)w * (2 * 2560);
    {   // prologue: stage K-chunk 0
        char* l = mybuf + (size_t)lane * 80;
        xasync16(xrow + 0,  l + 0);
        xasync16(xrow + 4,  l + 16);
        xasync16(xrow + 16, l + 32);
        xasync16(xrow + 20, l + 48);
    }
#endif

#pragma unroll 1
    for (int dc = 0; dc < 16; ++dc, bp += 2048) {
        float4 f0, f1, f2, f3;
#if HAS_ASYNC
        if (dc < 15) {   // stage next chunk, then wait only for the older batch
            const float* g = xrow + (dc + 1) * 32;
            char* l = mybuf + ((dc + 1) & 1) * 2560 + (size_t)lane * 80;
            xasync16(g + 0,  l + 0);
            xasync16(g + 4,  l + 16);
            xasync16(g + 16, l + 32);
            xasync16(g + 20, l + 48);
            __builtin_amdgcn_s_wait_asynccnt(4);
        } else {
            __builtin_amdgcn_s_wait_asynccnt(0);
        }
        const float4* sb = (const float4*)(mybuf + (dc & 1) * 2560 + (size_t)lane * 80);
        f0 = sb[0]; f1 = sb[1]; f2 = sb[2]; f3 = sb[3];   // ds_load_b128 x4
#else
        const float* xp = xrow + dc * 32;
        f0 = *(const float4*)(xp + 0);
        f1 = *(const float4*)(xp + 4);
        f2 = *(const float4*)(xp + 16);
        f3 = *(const float4*)(xp + 20);
        if (dc < 15) __builtin_prefetch(xp + 32, 0, 1);
#endif
        float f[16] = {f0.x, f0.y, f0.z, f0.w, f1.x, f1.y, f1.z, f1.w,
                       f2.x, f2.y, f2.z, f2.w, f3.x, f3.y, f3.z, f3.w};
        v16bf ahi, alo;
        float s = 0.f;
#pragma unroll
        for (int i = 0; i < 16; ++i) {
            s += f[i] * f[i];
            __bf16 h = (__bf16)f[i];
            ahi[i] = h;
            alo[i] = (__bf16)(f[i] - (float)h);
        }
        xsq += s;

        // ---- tile pairs: explicit 2-deep load pipeline, fenced per pair ----
        v16bf cb0 = *(const v16bf*)(bp + 0);
        v16bf cb1 = *(const v16bf*)(bp + 1024);
        v16bf cb2 = *(const v16bf*)(bp + 32768);
        v16bf cb3 = *(const v16bf*)(bp + 32768 + 1024);
        v16bf nb0 = cb0, nb1 = cb1, nb2 = cb2, nb3 = cb3;
#pragma unroll
        for (int p = 0; p < 8; ++p) {
            if (p < 7) {   // issue next pair's 8-load clause before this pair's math
                const char* nbp = bp + (size_t)(2 * p + 2) * 32768;
                nb0 = *(const v16bf*)(nbp + 0);
                nb1 = *(const v16bf*)(nbp + 1024);
                nb2 = *(const v16bf*)(nbp + 32768);
                nb3 = *(const v16bf*)(nbp + 32768 + 1024);
            }
            const int ct = 2 * p;
            acc[ct]     = __builtin_amdgcn_wmma_f32_16x16x32_bf16(
                false, ahi, false, cb0, (short)0, acc[ct],     false, false);
            acc[ct + 1] = __builtin_amdgcn_wmma_f32_16x16x32_bf16(
                false, ahi, false, cb2, (short)0, acc[ct + 1], false, false);
            acc[ct]     = __builtin_amdgcn_wmma_f32_16x16x32_bf16(
                false, ahi, false, cb1, (short)0, acc[ct],     false, false);
            acc[ct + 1] = __builtin_amdgcn_wmma_f32_16x16x32_bf16(
                false, ahi, false, cb3, (short)0, acc[ct + 1], false, false);
            acc[ct]     = __builtin_amdgcn_wmma_f32_16x16x32_bf16(
                false, alo, false, cb0, (short)0, acc[ct],     false, false);
            acc[ct + 1] = __builtin_amdgcn_wmma_f32_16x16x32_bf16(
                false, alo, false, cb2, (short)0, acc[ct + 1], false, false);
            SCHED_FENCE();   // cap in-flight B data at 2 pairs -> no spills
            cb0 = nb0; cb1 = nb1; cb2 = nb2; cb3 = nb3;
        }
    }

    // ---- epilogue: d2 -> student-t -> row-normalize -------------------------
    float xsqrow = xsq + __shfl_xor(xsq, 16);

    float xs[8];
#pragma unroll
    for (int j = 0; j < 8; ++j)
        xs[j] = __shfl(xsqrow, j + laneHi * 8);   // row rg*16 + j + laneHi*8

    float csv[16];
#pragma unroll
    for (int ct = 0; ct < 16; ++ct) csv[ct] = csq[cg * 256 + ct * 16 + ln];

    float p[8];
#pragma unroll
    for (int j = 0; j < 8; ++j) p[j] = 0.f;

#pragma unroll
    for (int ct = 0; ct < 16; ++ct)
#pragma unroll
        for (int j = 0; j < 8; ++j) {
            float d2 = xs[j] + csv[ct] - 2.f * acc[ct][j];
            float q  = 1.f / (1.f + d2);
            acc[ct][j] = q;
            p[j] += q;
        }

#pragma unroll
    for (int j = 0; j < 8; ++j) {
        float v = p[j];
        v += __shfl_xor(v, 1);
        v += __shfl_xor(v, 2);
        v += __shfl_xor(v, 4);
        v += __shfl_xor(v, 8);
        p[j] = v;
    }

    if (ln == 0) {
#pragma unroll
        for (int j = 0; j < 8; ++j)
            atomicAdd(&rowsum[rg * 16 + j + laneHi * 8], p[j]);  // ds_add_f32
    }
    __syncthreads();

    float inv[8];
#pragma unroll
    for (int j = 0; j < 8; ++j)
        inv[j] = 1.f / rowsum[rg * 16 + j + laneHi * 8];

#pragma unroll
    for (int ct = 0; ct < 16; ++ct)
#pragma unroll
        for (int j = 0; j < 8; ++j) {
            size_t row = (size_t)(blockRow + rg * 16 + j + laneHi * 8);
            out[row * KC + cg * 256 + ct * 16 + ln] = acc[ct][j] * inv[j];
        }
}

extern "C" void kernel_launch(void* const* d_in, const int* in_sizes, int n_in,
                              void* d_out, int out_size, void* d_ws, size_t ws_size,
                              hipStream_t stream) {
    const float* x = (const float*)d_in[0];       // [65536, 512] f32
    const float* c = (const float*)d_in[1];       // [512, 512]   f32
    float* out = (float*)d_out;                   // [65536, 512] f32
    char*  wsB = (char*)d_ws;                     // 1 MB fragments
    float* csq = (float*)((char*)d_ws + (1u << 20));

    prep_b  <<<512, 32,  0, stream>>>(c, wsB);
    prep_csq<<<64,  256, 0, stream>>>(c, csq);
    qkernel <<<NROWS / BLK_M, 512, 0, stream>>>(x, wsB, csq, out);
}
